// Correlation_73194832658666
// MI455X (gfx1250) — matrix operand order repaired and checked by reading, not compiled, not run on hardware
//
#include <hip/hip_runtime.h>

typedef __attribute__((ext_vector_type(16))) _Float16 v16h;
typedef __attribute__((ext_vector_type(8)))  _Float16 v8h;
typedef __attribute__((ext_vector_type(4)))  _Float16 v4h;
typedef __attribute__((ext_vector_type(8)))  float    v8f;

// Problem constants (from reference): N=8, C=256, H=64, W=128, 81 displacements.
#define NB    8
#define CC    256
#define HH    64
#define WW    128
#define DISP  4
#define NDISP 81

// LDS layout: [position][channel] in f16, pitch 264 halves = 528 bytes.
// 528 B is 16B-aligned (b128 fragment loads) and 132 dwords (=4 mod 64) keeps
// DS bank conflicts to ~2-way on wave32 b128 reads.
#define PITCH    264                 // halves per position
#define IN2_POS  (9 * 32)            // 9 dy rows x 32 columns (x0-4 .. x0+27)
#define IN1_BASE (IN2_POS * PITCH)   // 76032 halves
#define SMEM_HALVES ((IN2_POS + 16) * PITCH)  // 80256 halves = 160512 B

__global__ __launch_bounds__(256)
void corr_wmma_kernel(const float* __restrict__ in1,
                      const float* __restrict__ in2,
                      float* __restrict__ out) {
    __shared__ __attribute__((aligned(16))) _Float16 smem[SMEM_HALVES];

    const int tid = threadIdx.x;
    const int bid = blockIdx.x;
    const int xt = bid & 7;            // 8 x-strips of 16
    const int y  = (bid >> 3) & 63;    // H
    const int n  = bid >> 9;           // N
    const int x0 = xt << 4;

    // ---------------- Fill phase: global f32 -> f16 LDS ----------------
    // in2 tile: 288 positions x 256 channels, as 18432 channel-quads.
    // 256 threads -> 72 iterations; lanes map to consecutive x (coalesced).
    #pragma unroll 4
    for (int i = 0; i < 72; ++i) {
        int q   = i * 256 + tid;
        int col = q & 31;
        int r   = (q >> 5) % 9;
        int cq  = q / 288;
        int c   = cq << 2;
        int ygl = y - DISP + r;
        int xgl = x0 - DISP + col;
        float f0 = 0.f, f1 = 0.f, f2 = 0.f, f3 = 0.f;
        if ((unsigned)ygl < (unsigned)HH && (unsigned)xgl < (unsigned)WW) {
            const float* gp = in2 + (((n * CC + c) * HH + ygl) * WW + xgl);
            f0 = gp[0];
            f1 = gp[HH * WW];
            f2 = gp[2 * HH * WW];
            f3 = gp[3 * HH * WW];
        }
        v4h hv = { (_Float16)f0, (_Float16)f1, (_Float16)f2, (_Float16)f3 };
        *(v4h*)(&smem[(r * 32 + col) * PITCH + c]) = hv;   // ds_store_b64
    }
    // in1 tile: 16 positions x 256 channels = 1024 quads -> 4 iterations.
    #pragma unroll
    for (int i = 0; i < 4; ++i) {
        int q   = i * 256 + tid;
        int col = q & 15;
        int c   = (q >> 4) << 2;
        const float* gp = in1 + (((n * CC + c) * HH + y) * WW + x0 + col);
        v4h hv = { (_Float16)gp[0], (_Float16)gp[HH * WW],
                   (_Float16)gp[2 * HH * WW], (_Float16)gp[3 * HH * WW] };
        *(v4h*)(&smem[IN1_BASE + col * PITCH + c]) = hv;
    }
    __syncthreads();

    // ---------------- Compute phase: banded GEMM via WMMA ----------------
    const int wv    = tid >> 5;
    const int lane  = tid & 31;
    const int lhalf = lane >> 4;   // 0: lanes 0-15, 1: lanes 16-31
    const int lmod  = lane & 15;

    // A fragments are invariant across all jobs of this wave: build once.
    // Row M = lmod; K chunks [0..7,16..23] (lanes<16) or [8..15,24..31]
    // (lanes>=16) per the 16-bit A VGPR layout.
    const _Float16* aBase = &smem[IN1_BASE + lmod * PITCH + (lhalf ? 8 : 0)];
    v16h aF[8];
    #pragma unroll
    for (int kk = 0; kk < 8; ++kk) {
        v8h a_lo = *(const v8h*)(aBase + kk * 32);       // ds_load_b128
        v8h a_hi = *(const v8h*)(aBase + kk * 32 + 16);  // K +16 chunk
        aF[kk] = __builtin_shufflevector(a_lo, a_hi,
                     0,1,2,3,4,5,6,7,8,9,10,11,12,13,14,15);
    }

    // 18 jobs = 9 dy x 2 column tiles, round-robin over 8 waves.
    for (int job = wv; job < 18; job += 8) {
        const int dy = job >> 1;
        const int jt = job & 1;

        // B fragment source: col N = lmod of position (dy, jt*16 + lmod);
        // K [0..7,16..23] (lanes<16) or [8..15,24..31] (lanes>=16).
        const _Float16* bBase =
            &smem[(dy * 32 + jt * 16 + lmod) * PITCH + (lhalf ? 8 : 0)];

        // Preload ALL B fragments for this job into distinct registers so
        // the DS loads issue as a burst and the WMMA chain overlaps them
        // (staggered s_wait_dscnt instead of dscnt==0 before every WMMA).
        v8h bl[8], bh[8];
        #pragma unroll
        for (int kk = 0; kk < 8; ++kk) {
            bl[kk] = *(const v8h*)(bBase + kk * 32);      // ds_load_b128
            bh[kk] = *(const v8h*)(bBase + kk * 32 + 16); // ds_load_b128
        }

        v8f acc = {};
        #pragma unroll
        for (int kk = 0; kk < 8; ++kk) {
            v16h bF = __builtin_shufflevector(bl[kk], bh[kk],
                          0,1,2,3,4,5,6,7,8,9,10,11,12,13,14,15);
            acc = __builtin_amdgcn_wmma_f32_16x16x32_f16(
                      false, aF[kk], false, bF, (short)0, acc, false, false);
        }

        // Writeback: extract the 9-wide diagonal band.
        // D VGPR r: M = r + 8*lhalf, N = lmod. Column index = jt*16 + N.
        // dx+4 = (jt*16 + N) - M, valid in [0,8]. d = dy*9 + (dx+4).
        float* outBase = out + (((n * NDISP + dy * 9) * HH + y) * WW + x0);
        #pragma unroll
        for (int r = 0; r < 8; ++r) {
            const int m    = r + 8 * lhalf;
            const int dxp4 = jt * 16 + lmod - m;
            if ((unsigned)dxp4 <= 8u) {
                outBase[dxp4 * (HH * WW) + m] = acc[r] * 0.00390625f; // 1/256
            }
        }
    }
}

extern "C" void kernel_launch(void* const* d_in, const int* in_sizes, int n_in,
                              void* d_out, int out_size, void* d_ws, size_t ws_size,
                              hipStream_t stream) {
    (void)in_sizes; (void)n_in; (void)out_size; (void)d_ws; (void)ws_size;
    const float* in1 = (const float*)d_in[0];
    const float* in2 = (const float*)d_in[1];
    float* out = (float*)d_out;
    dim3 grid(NB * HH * (WW / 16));   // 8 * 64 * 8 = 4096 workgroups
    dim3 block(256);                  // 8 wave32
    corr_wmma_kernel<<<grid, block, 0, stream>>>(in1, in2, out);
}
